// GDEncoder_70858370450079
// MI455X (gfx1250) — compile-verified, matrix operand order checked by reading, not run on hardware
//
#include <hip/hip_runtime.h>
#include <hip/hip_bf16.h>

typedef __attribute__((ext_vector_type(16))) _Float16 v16h;
typedef __attribute__((ext_vector_type(8)))  float    v8f;
typedef __attribute__((address_space(3))) _Float16 lds_f16_t;

// ---------------------------------------------------------------------------
// WMMA fragment helpers (CDNA5 16x16x32 f16, wave32)
// A (16x32, MxK) / B (32x16 as NxK rows): lane L holds row/col L%16;
// halves j<8 -> K = hi*8 + j ; halves j>=8 -> K = 16 + hi*8 + (j-8), hi = L>=16.
// Source must be row-major with K contiguous; 16B alignment per half-row chunk.
// ---------------------------------------------------------------------------
__device__ __forceinline__ v16h load_frag_f16(const _Float16* base, int stride) {
  const int lane = threadIdx.x & 31;
  const int row  = lane & 15;
  const int hi   = lane >> 4;
  const _Float16* p = base + row * stride + hi * 8;
  v16h f;
  *((float4*)&f)       = *(const float4*)(p);        // K = hi*8 + 0..7
  *(((float4*)&f) + 1) = *(const float4*)(p + 16);   // K = 16 + hi*8 + 0..7
  return f;
}

// C/D f32 16x16: lane L: col = L%16, rows (L>=16 ? 8 : 0) + v, v=0..7
__device__ __forceinline__ void store_cd_f32(float* base, int stride, v8f c) {
  const int lane = threadIdx.x & 31;
  const int col  = lane & 15;
  const int r0   = (lane >> 4) * 8;
#pragma unroll
  for (int v = 0; v < 8; ++v) base[(r0 + v) * stride + col] = c[v];
}
__device__ __forceinline__ void store_cd_f16(_Float16* base, int stride, v8f c) {
  const int lane = threadIdx.x & 31;
  const int col  = lane & 15;
  const int r0   = (lane >> 4) * 8;
#pragma unroll
  for (int v = 0; v < 8; ++v) base[(r0 + v) * stride + col] = (_Float16)c[v];
}
// generic (r,c) strided store (used for transposed V)
__device__ __forceinline__ void store_cd_f16_rc(_Float16* base, long rs, long cs, v8f c) {
  const int lane = threadIdx.x & 31;
  const int col  = lane & 15;
  const int r0   = (lane >> 4) * 8;
#pragma unroll
  for (int v = 0; v < 8; ++v) base[(long)(r0 + v) * rs + (long)col * cs] = (_Float16)c[v];
}
__device__ __forceinline__ v8f bias_frag(const float* bias /* already + nt*16 */) {
  const float b = bias[threadIdx.x & 15];
  v8f c;
#pragma unroll
  for (int v = 0; v < 8; ++v) c[v] = b;
  return c;
}
__device__ __forceinline__ v8f wmma_f16(v16h a, v16h b, v8f c) {
  return __builtin_amdgcn_wmma_f32_16x16x32_f16(false, a, false, b, (short)0, c,
                                                false, false);
}
__device__ __forceinline__ float sigmoidf(float x) { return 1.0f / (1.0f + __expf(-x)); }

// async 16B global->LDS copy (gfx1250 VGLOBAL async encoding; ASYNCcnt-tracked)
__device__ __forceinline__ void async_copy_b128(void* lds_dst, const void* gsrc) {
  unsigned lds_off = (unsigned)(unsigned long)(lds_f16_t*)lds_dst;  // AS3 = LDS byte offset
  asm volatile("global_load_async_to_lds_b128 %0, %1, off"
               :: "v"(lds_off), "v"(gsrc) : "memory");
}
__device__ __forceinline__ void wait_async0() {
  asm volatile("s_wait_asynccnt 0x0" ::: "memory");
}

// ---------------------------------------------------------------------------
// fp32 -> fp16 weight conversion
// ---------------------------------------------------------------------------
__global__ void cvt_f16_kernel(const float* __restrict__ src, _Float16* __restrict__ dst, int n) {
  int i = blockIdx.x * blockDim.x + threadIdx.x;
  if (i < n) dst[i] = (_Float16)src[i];
}

// ---------------------------------------------------------------------------
// elu(x @ W1^T + b1): x (rows,8) f32 -> y (rows,32) f16
// ---------------------------------------------------------------------------
__global__ void prelinear_elu_kernel(const float* __restrict__ x, const float* __restrict__ W,
                                     const float* __restrict__ b, _Float16* __restrict__ y,
                                     int rows) {
  int i = blockIdx.x * blockDim.x + threadIdx.x;
  if (i >= rows * 32) return;
  int r = i >> 5, c = i & 31;
  const float* xr = x + (long)r * 8;
  float acc = b[c];
#pragma unroll
  for (int k = 0; k < 8; ++k) acc += xr[k] * W[c * 8 + k];
  y[i] = (_Float16)(acc > 0.0f ? acc : (__expf(acc) - 1.0f));
}

// ---------------------------------------------------------------------------
// behavior pre-linear: BLE (16,64,8,6) -> leaky_relu -> beh_x (64,16,128) f16
// ---------------------------------------------------------------------------
__global__ void beh_prelinear_kernel(const float* __restrict__ ble, const float* __restrict__ Wb,
                                     const float* __restrict__ bb, _Float16* __restrict__ y) {
  int e = blockIdx.x * blockDim.x + threadIdx.x;
  if (e >= 64 * 16 * 8 * 16) return;
  int c = e & 15, car = (e >> 4) & 7, t = (e >> 7) & 15, b = e >> 11;
  const float* src = ble + (((long)t * 64 + b) * 8 + car) * 6;
  float acc = bb[c];
#pragma unroll
  for (int k = 0; k < 6; ++k) acc += src[k] * Wb[c * 6 + k];
  y[((long)b * 16 + t) * 128 + car * 16 + c] = (_Float16)(acc > 0.0f ? acc : 0.1f * acc);
}

// ---------------------------------------------------------------------------
// LSTM (torch gate order i,f,g,o). One workgroup = one batch chunk of MT*16
// rows; 8 waves; per step WMMA computes z = x@Wih^T + h@Whh^T + b into LDS,
// then elementwise gates update h (f16 in LDS, feeds next step) and c (f32).
// Output element (s, row, h) written to out + s*ss + row*sm + h.
// ---------------------------------------------------------------------------
template <int F, int MT>
__global__ void __launch_bounds__(256)
lstm_kernel(const _Float16* __restrict__ x,      // (S, batch_total, F)
            const _Float16* __restrict__ Wih,    // (256, F) f16
            const _Float16* __restrict__ Whh,    // (256, 64) f16
            const float* __restrict__ bias,      // (256) f32
            _Float16* __restrict__ out, int S, int batch_total, long ss, long sm) {
  constexpr int CHUNK = MT * 16;
  constexpr int TILES = MT * 16;        // MT m-tiles x 16 n-tiles
  constexpr int TPW   = TILES / 8;
  __shared__ __align__(16) _Float16 lds_h[CHUNK * 64];
  __shared__ __align__(16) float    lds_c[CHUNK * 64];
  __shared__ __align__(16) float    lds_z[CHUNK * 256];
  const int tid  = threadIdx.x;
  const int wave = tid >> 5;
  const long rowbase = (long)blockIdx.x * CHUNK;

  for (int i = tid; i < CHUNK * 64; i += 256) { lds_h[i] = (_Float16)0.0f; lds_c[i] = 0.0f; }
  __syncthreads();

  for (int s = 0; s < S; ++s) {
    const _Float16* xs = x + ((long)s * batch_total + rowbase) * F;
#pragma unroll
    for (int i = 0; i < TPW; ++i) {
      int tile = wave * TPW + i;
      int mt = tile >> 4, nt = tile & 15;
      v8f acc = bias_frag(bias + nt * 16);
#pragma unroll
      for (int kk = 0; kk < F / 32; ++kk) {
        v16h a = load_frag_f16(xs + mt * 16 * F + kk * 32, F);
        v16h b = load_frag_f16(Wih + nt * 16 * F + kk * 32, F);
        acc = wmma_f16(a, b, acc);
      }
#pragma unroll
      for (int kk = 0; kk < 2; ++kk) {
        v16h a = load_frag_f16(lds_h + mt * 16 * 64 + kk * 32, 64);
        v16h b = load_frag_f16(Whh + nt * 16 * 64 + kk * 32, 64);
        acc = wmma_f16(a, b, acc);
      }
      store_cd_f32(lds_z + mt * 16 * 256 + nt * 16, 256, acc);
    }
    __syncthreads();
    for (int e = tid; e < CHUNK * 64; e += 256) {
      int m = e >> 6, hc = e & 63;
      float zi = lds_z[m * 256 + hc];
      float zf = lds_z[m * 256 + 64 + hc];
      float zg = lds_z[m * 256 + 128 + hc];
      float zo = lds_z[m * 256 + 192 + hc];
      float cc = sigmoidf(zf) * lds_c[e] + sigmoidf(zi) * tanhf(zg);
      float hh = sigmoidf(zo) * tanhf(cc);
      lds_c[e] = cc;
      lds_h[e] = (_Float16)hh;
      out[(long)s * ss + (rowbase + m) * sm + hc] = (_Float16)hh;
    }
    __syncthreads();
  }
}

// ---------------------------------------------------------------------------
// QKV projection: nb_h (26624,64) f16 -> q,k (16,1664,64) f16 and vT (16,64,1664)
// One WG per 64 rows (blocks never straddle t since 1664 % 64 == 0).
// ---------------------------------------------------------------------------
__global__ void __launch_bounds__(256)
qkv_kernel(const _Float16* __restrict__ nbh,
           const _Float16* __restrict__ Wq, const float* __restrict__ bq,
           const _Float16* __restrict__ Wk, const float* __restrict__ bk,
           const _Float16* __restrict__ Wv, const float* __restrict__ bv,
           _Float16* __restrict__ q, _Float16* __restrict__ k, _Float16* __restrict__ vT) {
  const int wave = threadIdx.x >> 5;
  const long rowbase = (long)blockIdx.x * 64;
  const int t  = (int)(rowbase / 1664);
  const int kb = (int)(rowbase % 1664);
#pragma unroll
  for (int i = 0; i < 6; ++i) {
    int tile = wave * 6 + i;            // 48 tiles: 3 proj x 4 mt x 4 nt
    int which = tile >> 4;
    int rem = tile & 15;
    int mt = rem >> 2, nt = rem & 3;
    const _Float16* Ws = (which == 0) ? Wq : ((which == 1) ? Wk : Wv);
    const float*    bs = (which == 0) ? bq : ((which == 1) ? bk : bv);
    v8f acc = bias_frag(bs + nt * 16);
#pragma unroll
    for (int kk = 0; kk < 2; ++kk) {
      v16h a = load_frag_f16(nbh + (rowbase + mt * 16) * 64 + kk * 32, 64);
      v16h b = load_frag_f16(Ws + nt * 16 * 64 + kk * 32, 64);
      acc = wmma_f16(a, b, acc);
    }
    if (which == 0)
      store_cd_f16(q + (rowbase + mt * 16) * 64 + nt * 16, 64, acc);
    else if (which == 1)
      store_cd_f16(k + (rowbase + mt * 16) * 64 + nt * 16, 64, acc);
    else  // v stored transposed: vT[(t*64 + feat)*1664 + key]
      store_cd_f16_rc(vT + ((long)t * 64 + nt * 16) * 1664 + kb + mt * 16,
                      /*rs=*/1, /*cs=*/1664, acc);
  }
}

// ---------------------------------------------------------------------------
// Flash attention per (t, 64-query block). Online softmax; O in wave registers.
// Q block staged into LDS via gfx1250 async global->LDS DMA (ASYNCcnt);
// next K / V^T blocks prefetched into GL2 while the current one is consumed.
// ---------------------------------------------------------------------------
__global__ void __launch_bounds__(256)
attn_kernel(const _Float16* __restrict__ q, const _Float16* __restrict__ k,
            const _Float16* __restrict__ vT, float* __restrict__ o) {
  __shared__ __align__(16) _Float16 lds_q[64 * 64];
  __shared__ __align__(16) float    lds_s[64 * 64];
  __shared__ __align__(16) _Float16 lds_p[64 * 64];
  __shared__ float lds_scale[64];
  const int tid  = threadIdx.x;
  const int wave = tid >> 5;
  const int t  = blockIdx.y;
  const int qb = blockIdx.x * 64;

  // async-stage Q block (64x64 f16 = 8KB): 512 x 16B chunks, 2 per thread
  const _Float16* gq = q + ((long)t * 1664 + qb) * 64;
#pragma unroll
  for (int i = 0; i < 2; ++i) {
    int chunk = tid + i * 256;          // 0..511
    async_copy_b128(&lds_q[chunk * 8], gq + chunk * 8);
  }
  wait_async0();
  __syncthreads();

  float m_i = -3.0e38f, l_i = 0.0f;   // valid in lanes tid<64
  v8f o0 = {}, o1 = {};

  for (int kbk = 0; kbk < 1664; kbk += 64) {
    // prefetch next K / V^T blocks into GL2 (one byte per lane = full lines)
    if (kbk + 64 < 1664) {
      __builtin_prefetch(k + ((long)t * 1664 + kbk + 64) * 64 + tid * 16, 0, 3);
      __builtin_prefetch(vT + ((long)t * 64 + (tid >> 2)) * 1664 + kbk + 64 + (tid & 3) * 16, 0, 3);
    }
    // S = Q Kblk^T, 16 tiles, 2 per wave
#pragma unroll
    for (int i = 0; i < 2; ++i) {
      int tile = wave * 2 + i;
      int mt = tile >> 2, nt = tile & 3;
      v8f acc = {};
#pragma unroll
      for (int kk = 0; kk < 2; ++kk) {
        v16h a = load_frag_f16(lds_q + mt * 16 * 64 + kk * 32, 64);
        v16h b = load_frag_f16(k + ((long)t * 1664 + kbk + nt * 16) * 64 + kk * 32, 64);
        acc = wmma_f16(a, b, acc);
      }
      store_cd_f32(lds_s + mt * 16 * 64 + nt * 16, 64, acc);
    }
    __syncthreads();
    if (tid < 64) {
      const int r = tid;
      float mx = -3.0e38f;
      for (int j = 0; j < 64; ++j) mx = fmaxf(mx, lds_s[r * 64 + j] * 0.125f);
      float mn = fmaxf(m_i, mx);
      float sc = __expf(m_i - mn);
      float sum = 0.0f;
      for (int j = 0; j < 64; ++j) {
        float p = __expf(lds_s[r * 64 + j] * 0.125f - mn);
        lds_p[r * 64 + j] = (_Float16)p;
        sum += p;
      }
      l_i = l_i * sc + sum;
      m_i = mn;
      lds_scale[r] = sc;
    }
    __syncthreads();
    // O = O*scale + P @ Vblk
    {
      const int r0 = ((tid & 31) >> 4) * 8;
#pragma unroll
      for (int i = 0; i < 2; ++i) {
        int tile = wave * 2 + i;
        int mt = tile >> 2, nt = tile & 3;
        v8f acc = (i == 0) ? o0 : o1;
#pragma unroll
        for (int v = 0; v < 8; ++v) acc[v] *= lds_scale[mt * 16 + r0 + v];
#pragma unroll
        for (int kk = 0; kk < 2; ++kk) {
          v16h a = load_frag_f16(lds_p + mt * 16 * 64 + kk * 32, 64);
          v16h b = load_frag_f16(vT + ((long)t * 64 + nt * 16) * 1664 + kbk + kk * 32, 1664);
          acc = wmma_f16(a, b, acc);
        }
        if (i == 0) o0 = acc; else o1 = acc;
      }
    }
    __syncthreads();
  }
  if (tid < 64) lds_scale[tid] = 1.0f / l_i;
  __syncthreads();
  {
    const int lane = tid & 31;
    const int col = lane & 15;
    const int r0 = (lane >> 4) * 8;
#pragma unroll
    for (int i = 0; i < 2; ++i) {
      int tile = wave * 2 + i;
      int mt = tile >> 2, nt = tile & 3;
      v8f acc = (i == 0) ? o0 : o1;
      float* base = o + ((long)t * 1664 + qb + mt * 16) * 64 + nt * 16;
#pragma unroll
      for (int v = 0; v < 8; ++v)
        base[(r0 + v) * 64 + col] = acc[v] * lds_scale[mt * 16 + r0 + v];
    }
  }
}

// ---------------------------------------------------------------------------
// mask running-index scan (reference cumsum trick) + per-sample 1/count
// ---------------------------------------------------------------------------
__global__ void mask_scan_kernel(const unsigned char* __restrict__ mask,
                                 int* __restrict__ idx, float* __restrict__ inv_cnt) {
  if (threadIdx.x != 0 || blockIdx.x != 0) return;
  int run = 0;
  for (int b = 0; b < 64; ++b) {
    int cnt = 0;
    for (int g = 0; g < 39; ++g) {
      int i = b * 39 + g;
      int m = mask[i] ? 1 : 0;
      run += m;
      cnt += m;
      int v = run - 1;
      if (v < 0) v = 0;
      if (v > 1663) v = 1663;
      idx[i] = m ? v : -1;
    }
    inv_cnt[b] = 1.0f / (float)(cnt < 1 ? 1 : cnt);
  }
}

// pooled[b,t,h] = mean over valid grid cells -> cat cols 64..127 (f16)
__global__ void __launch_bounds__(64)
pool_kernel(const float* __restrict__ att, const int* __restrict__ idx,
            const float* __restrict__ inv_cnt, _Float16* __restrict__ cat) {
  const int bt = blockIdx.x;          // b*16 + t
  const int b = bt >> 4, t = bt & 15;
  const int h = threadIdx.x;
  float acc = 0.0f;
  for (int g = 0; g < 39; ++g) {
    int ix = idx[b * 39 + g];
    if (ix >= 0) acc += att[((long)t * 1664 + ix) * 64 + h];
  }
  cat[(long)bt * 192 + 64 + h] = (_Float16)(acc * inv_cnt[b]);
}

// ---------------------------------------------------------------------------
// fuse: cat (1024,192) @ Wp^T (384,192) + bp -> outp (1024,384) f16
// ---------------------------------------------------------------------------
__global__ void __launch_bounds__(256)
fuse_kernel(const _Float16* __restrict__ cat, const _Float16* __restrict__ Wp,
            const float* __restrict__ bp, _Float16* __restrict__ outp) {
  const int wave = threadIdx.x >> 5;
  const long rowbase = (long)blockIdx.x * 64;
#pragma unroll
  for (int i = 0; i < 12; ++i) {
    int tile = wave * 12 + i;           // 96 tiles: 4 mt x 24 nt
    int mt = tile / 24, nt = tile % 24;
    v8f acc = bias_frag(bp + nt * 16);
#pragma unroll
    for (int kk = 0; kk < 6; ++kk) {
      v16h a = load_frag_f16(cat + (rowbase + mt * 16) * 192 + kk * 32, 192);
      v16h b = load_frag_f16(Wp + nt * 16 * 192 + kk * 32, 192);
      acc = wmma_f16(a, b, acc);
    }
    store_cd_f16(outp + (rowbase + mt * 16) * 384 + nt * 16, 384, acc);
  }
}

// ---------------------------------------------------------------------------
// GLU: out = (outp@Wa^T+ba) * sigmoid(outp@Wg^T+bg) -> d_out f32 (1024,64)
// Each wave computes both a/g accumulators for its tiles; combine in regs.
// ---------------------------------------------------------------------------
__global__ void __launch_bounds__(256)
glu_kernel(const _Float16* __restrict__ outp,
           const _Float16* __restrict__ Wa, const float* __restrict__ ba,
           const _Float16* __restrict__ Wg, const float* __restrict__ bg,
           float* __restrict__ out) {
  const int wave = threadIdx.x >> 5;
  const long rowbase = (long)blockIdx.x * 64;
  const int lane = threadIdx.x & 31;
  const int col = lane & 15, r0 = (lane >> 4) * 8;
#pragma unroll
  for (int i = 0; i < 2; ++i) {
    int tile = wave * 2 + i;            // 16 tiles: 4 mt x 4 nt
    int mt = tile >> 2, nt = tile & 3;
    v8f acca = bias_frag(ba + nt * 16);
    v8f accg = bias_frag(bg + nt * 16);
#pragma unroll
    for (int kk = 0; kk < 12; ++kk) {
      v16h a  = load_frag_f16(outp + (rowbase + mt * 16) * 384 + kk * 32, 384);
      v16h bA = load_frag_f16(Wa + nt * 16 * 384 + kk * 32, 384);
      v16h bG = load_frag_f16(Wg + nt * 16 * 384 + kk * 32, 384);
      acca = wmma_f16(a, bA, acca);
      accg = wmma_f16(a, bG, accg);
    }
    float* base = out + (rowbase + mt * 16) * 64 + nt * 16;
#pragma unroll
    for (int v = 0; v < 8; ++v)
      base[(r0 + v) * 64 + col] = acca[v] * sigmoidf(accg[v]);
  }
}

// ---------------------------------------------------------------------------
// host side
// ---------------------------------------------------------------------------
extern "C" void kernel_launch(void* const* d_in, const int* in_sizes, int n_in,
                              void* d_out, int out_size, void* d_ws, size_t ws_size,
                              hipStream_t stream) {
  (void)in_sizes; (void)n_in; (void)out_size; (void)ws_size;
  const float* hist = (const float*)d_in[0];
  const float* nbrs = (const float*)d_in[1];
  const float* ble  = (const float*)d_in[2];
  const unsigned char* mask = (const unsigned char*)d_in[3];
  const float* W1 = (const float*)d_in[4];  const float* b1 = (const float*)d_in[5];
  const float* Wb = (const float*)d_in[6];  const float* bb = (const float*)d_in[7];
  const float* Wih = (const float*)d_in[8]; const float* Whh = (const float*)d_in[9];
  const float* bl  = (const float*)d_in[10];
  const float* Wih2 = (const float*)d_in[11]; const float* Whh2 = (const float*)d_in[12];
  const float* bl2  = (const float*)d_in[13];
  const float* Wq = (const float*)d_in[14]; const float* bq = (const float*)d_in[15];
  const float* Wk = (const float*)d_in[16]; const float* bk = (const float*)d_in[17];
  const float* Wv = (const float*)d_in[18]; const float* bv = (const float*)d_in[19];
  const float* Wp = (const float*)d_in[20]; const float* bp = (const float*)d_in[21];
  const float* Wa = (const float*)d_in[22]; const float* ba = (const float*)d_in[23];
  const float* Wg = (const float*)d_in[24]; const float* bg = (const float*)d_in[25];
  float* outF = (float*)d_out;

  // workspace carve-out (256B aligned slabs)
  char* w = (char*)d_ws;
  auto alloc = [&](size_t bytes) -> char* {
    char* p = w;
    w += (bytes + 255) & ~(size_t)255;
    return p;
  };
  _Float16* Wih16  = (_Float16*)alloc(256 * 32 * 2);
  _Float16* Whh16  = (_Float16*)alloc(256 * 64 * 2);
  _Float16* Wih216 = (_Float16*)alloc(256 * 128 * 2);
  _Float16* Whh216 = (_Float16*)alloc(256 * 64 * 2);
  _Float16* Wq16   = (_Float16*)alloc(64 * 64 * 2);
  _Float16* Wk16   = (_Float16*)alloc(64 * 64 * 2);
  _Float16* Wv16   = (_Float16*)alloc(64 * 64 * 2);
  _Float16* Wp16   = (_Float16*)alloc(384 * 192 * 2);
  _Float16* Wa16   = (_Float16*)alloc(64 * 384 * 2);
  _Float16* Wg16   = (_Float16*)alloc(64 * 384 * 2);
  _Float16* xh_ego = (_Float16*)alloc((size_t)16 * 64 * 32 * 2);
  _Float16* xh_nbr = (_Float16*)alloc((size_t)16 * 1664 * 32 * 2);
  _Float16* beh_x  = (_Float16*)alloc((size_t)64 * 16 * 128 * 2);
  _Float16* nb_h   = (_Float16*)alloc((size_t)16 * 1664 * 64 * 2);
  _Float16* qbuf   = (_Float16*)alloc((size_t)16 * 1664 * 64 * 2);
  _Float16* kbuf   = (_Float16*)alloc((size_t)16 * 1664 * 64 * 2);
  _Float16* vTbuf  = (_Float16*)alloc((size_t)16 * 64 * 1664 * 2);
  float*    nb_att = (float*)alloc((size_t)16 * 1664 * 64 * 4);
  _Float16* catb   = (_Float16*)alloc((size_t)1024 * 192 * 2);
  _Float16* outp   = (_Float16*)alloc((size_t)1024 * 384 * 2);
  int*      idxb   = (int*)alloc(2496 * 4);
  float*    invc   = (float*)alloc(64 * 4);

  auto cvt = [&](const float* s, _Float16* d, int n) {
    cvt_f16_kernel<<<(n + 255) / 256, 256, 0, stream>>>(s, d, n);
  };
  cvt(Wih, Wih16, 256 * 32);
  cvt(Whh, Whh16, 256 * 64);
  cvt(Wih2, Wih216, 256 * 128);
  cvt(Whh2, Whh216, 256 * 64);
  cvt(Wq, Wq16, 64 * 64);
  cvt(Wk, Wk16, 64 * 64);
  cvt(Wv, Wv16, 64 * 64);
  cvt(Wp, Wp16, 384 * 192);
  cvt(Wa, Wa16, 64 * 384);
  cvt(Wg, Wg16, 64 * 384);

  // pre-linears
  prelinear_elu_kernel<<<(1024 * 32 + 255) / 256, 256, 0, stream>>>(hist, W1, b1, xh_ego, 1024);
  prelinear_elu_kernel<<<(26624 * 32 + 255) / 256, 256, 0, stream>>>(nbrs, W1, b1, xh_nbr, 26624);
  beh_prelinear_kernel<<<(131072 + 255) / 256, 256, 0, stream>>>(ble, Wb, bb, beh_x);

  // LSTMs (outputs written as f16 into cat slots / nb_h)
  lstm_kernel<32, 4><<<1, 256, 0, stream>>>(xh_ego, Wih16, Whh16, bl,
                                            catb, 16, 64, (long)192, (long)3072);
  lstm_kernel<32, 4><<<26, 256, 0, stream>>>(xh_nbr, Wih16, Whh16, bl,
                                             nb_h, 16, 1664, (long)1664 * 64, (long)64);
  lstm_kernel<128, 1><<<1, 256, 0, stream>>>(beh_x, Wih216, Whh216, bl2,
                                             catb + 128, 64, 16, (long)3072, (long)192);

  // attention
  qkv_kernel<<<416, 256, 0, stream>>>(nb_h, Wq16, bq, Wk16, bk, Wv16, bv, qbuf, kbuf, vTbuf);
  mask_scan_kernel<<<1, 32, 0, stream>>>(mask, idxb, invc);
  attn_kernel<<<dim3(26, 16), 256, 0, stream>>>(qbuf, kbuf, vTbuf, nb_att);
  pool_kernel<<<1024, 64, 0, stream>>>(nb_att, idxb, invc, catb);

  // fuse + GLU
  fuse_kernel<<<16, 256, 0, stream>>>(catb, Wp16, bp, outp);
  glu_kernel<<<16, 256, 0, stream>>>(outp, Wa16, ba, Wg16, bg, outF);
}
